// RNNBase_56049323213597
// MI455X (gfx1250) — matrix operand order repaired and checked by reading, not compiled, not run on hardware
//
#include <hip/hip_runtime.h>
#include <stdint.h>

// ---------------------------------------------------------------------------
// GRU + attention for MI455X (gfx1250), wave32, WMMA bf16 16x16x32.
//
//   k_f2bf        : x (fp32) -> xb (bf16)
//   k_transpose   : each weight W[K][N] fp32 -> WT[N][K] bf16 (B^T layout)
//   k_gemm_xg     : XG[g] = x @ W_x{z,r,h}; 2x4 register blocking, A staged
//                   in LDS via async-to-LDS, B^T streamed from L2
//   k_recurrent   : persistent 16-WG kernel, 512 serial steps, 2 grid
//                   barriers/step; A tile staged to LDS each phase
//   k_gemm_y      : outputs = Hs @ W_hq + b_q (2x4 blocked, LDS-staged A)
//   k_attn        : tanh energy, softmax over S, context
// ---------------------------------------------------------------------------

namespace {

constexpr int kB  = 32;    // batch
constexpr int kS  = 512;   // seq len
constexpr int kI  = 512;   // input dim
constexpr int kH  = 1024;  // hidden dim
constexpr int kM  = kB * kS;  // 16384 GEMM rows for batched parts
constexpr int kNWG = 16;      // persistent workgroups in recurrent kernel

// workspace layout (bytes, each section 64KB-multiple => aligned)
constexpr size_t SZ_XB  = (size_t)kM * kI * 2;
constexpr size_t SZ_WTX = 3ull * kH * kI * 2;
constexpr size_t SZ_WTH = 3ull * kH * kH * 2;
constexpr size_t SZ_WTQ = (size_t)kI * kH * 2;
constexpr size_t SZ_XG  = 3ull * kM * kH * 2;
constexpr size_t SZ_HS  = (size_t)kM * kH * 2;
constexpr size_t SZ_HF  = 2ull * kB * kH * 4;
constexpr size_t SZ_HB  = 2ull * kB * kH * 2;
constexpr size_t SZ_RH  = (size_t)kB * kH * 2;
constexpr size_t SZ_ZB  = (size_t)kB * kH * 4;
constexpr size_t OFF_XB  = 0;
constexpr size_t OFF_WTX = OFF_XB  + SZ_XB;
constexpr size_t OFF_WTH = OFF_WTX + SZ_WTX;
constexpr size_t OFF_WTQ = OFF_WTH + SZ_WTH;
constexpr size_t OFF_XG  = OFF_WTQ + SZ_WTQ;
constexpr size_t OFF_HS  = OFF_XG  + SZ_XG;
constexpr size_t OFF_HF  = OFF_HS  + SZ_HS;
constexpr size_t OFF_HB  = OFF_HF  + SZ_HF;
constexpr size_t OFF_RH  = OFF_HB  + SZ_HB;
constexpr size_t OFF_ZB  = OFF_RH  + SZ_RH;
constexpr size_t OFF_BAR = OFF_ZB  + SZ_ZB;

// output layout in d_out (floats)
constexpr size_t OUT_CTX = (size_t)kM * kI;
constexpr size_t OUT_ATT = OUT_CTX + (size_t)kB * kH;

typedef __bf16 bf16_t;
typedef bf16_t v8bf  __attribute__((ext_vector_type(8)));
typedef bf16_t v16bf __attribute__((ext_vector_type(16)));
typedef float  v8f   __attribute__((ext_vector_type(8)));

__device__ __forceinline__ unsigned short f2bf(float f) {
  union { float f; unsigned u; } v; v.f = f;
  unsigned r = (v.u + 0x7FFFu + ((v.u >> 16) & 1u)) >> 16;
  return (unsigned short)r;
}
__device__ __forceinline__ float bf2f(unsigned short h) {
  union { unsigned u; float f; } v; v.u = ((unsigned)h) << 16;
  return v.f;
}
__device__ __forceinline__ float sigmoidf(float x) {
  return 1.0f / (1.0f + expf(-x));
}

// A fragment from global, 16x32 bf16, row-major [M][K] (ld in elements).
__device__ __forceinline__ v16bf load_frag_a(const unsigned short* __restrict__ base,
                                             int ld, int mBase, int kBase, int lane) {
  int m  = mBase + (lane & 15);
  int ko = (lane >> 4) << 3;
  const bf16_t* p = reinterpret_cast<const bf16_t*>(base) + (size_t)m * ld + kBase + ko;
  v8bf lo = *reinterpret_cast<const v8bf*>(p);
  v8bf hi = *reinterpret_cast<const v8bf*>(p + 16);
  return __builtin_shufflevector(lo, hi, 0,1,2,3,4,5,6,7,8,9,10,11,12,13,14,15);
}

// B fragment: 32x16 bf16 from B^T stored row-major [N][K] (ld = K).
__device__ __forceinline__ v16bf load_frag_b(const unsigned short* __restrict__ base,
                                             int ld, int nBase, int kBase, int lane) {
  int n  = nBase + (lane & 15);
  int ko = (lane >> 4) << 3;
  const bf16_t* p = reinterpret_cast<const bf16_t*>(base) + (size_t)n * ld + kBase + ko;
  v8bf lo = *reinterpret_cast<const v8bf*>(p);
  v8bf hi = *reinterpret_cast<const v8bf*>(p + 16);
  return __builtin_shufflevector(lo, hi, 0,1,2,3,4,5,6,7,8,9,10,11,12,13,14,15);
}

__device__ __forceinline__ v8f wmma_bf16(v16bf a, v16bf b, v8f c) {
  return __builtin_amdgcn_wmma_f32_16x16x32_bf16(false, a, false, b, (short)0, c,
                                                 false, false);
}

// grid-wide sense barrier for kNWG co-resident workgroups
__device__ __forceinline__ void grid_barrier(int* cnt, volatile int* gen) {
  __threadfence();
  __syncthreads();
  if (threadIdx.x == 0) {
    int g = *gen;
    if (atomicAdd(cnt, 1) == kNWG - 1) {
      *cnt = 0;
      __threadfence();
      atomicAdd((int*)gen, 1);
    } else {
      while (*gen == g) { __builtin_amdgcn_s_sleep(2); }
    }
  }
  __syncthreads();
  __threadfence();
}

} // namespace

// ---- CDNA5 async global->LDS staging (guarded; falls back to plain copies) -
// clang's prototype (from round-2 diagnostic): param0 is int __vector(4) * in
// AS(1) ("__device__"), so cast to vector_size(16) int pointers, non-const.
typedef int v4i_vs __attribute__((vector_size(16)));

#if defined(__gfx1250__) && __has_builtin(__builtin_amdgcn_global_load_async_to_lds_b128)
#define ASYNC_COPY16(dst, src)                                                \
  __builtin_amdgcn_global_load_async_to_lds_b128(                             \
      (__attribute__((address_space(1))) v4i_vs*)(src),                       \
      (__attribute__((address_space(3))) v4i_vs*)(dst), 0, 0)
#if __has_builtin(__builtin_amdgcn_s_wait_asynccnt)
#define ASYNC_WAIT() __builtin_amdgcn_s_wait_asynccnt(0)
#else
#define ASYNC_WAIT() asm volatile("s_wait_asynccnt 0" ::: "memory")
#endif
#else
#define ASYNC_COPY16(dst, src) (*(uint4*)(dst) = *(const uint4*)(src))
#define ASYNC_WAIT() ((void)0)
#endif

// stage nChunks*16 contiguous bytes from global into shared array `smem`
#define STAGE_TO_LDS(smem, srcptr, nChunks)                                   \
  do {                                                                        \
    char* _s = (char*)(srcptr);                                               \
    for (int _c = threadIdx.x; _c < (nChunks); _c += 256)                     \
      ASYNC_COPY16(((char*)(smem)) + (size_t)_c * 16, _s + (size_t)_c * 16);  \
    ASYNC_WAIT();                                                             \
    __syncthreads();                                                          \
  } while (0)

// A fragment from an LDS-resident row-major tile (row = full local row index)
#define LOAD_FRAG_A_LDS(dstv, smem, ldk, mLocal, kBase, lane)                 \
  do {                                                                        \
    const bf16_t* _p = reinterpret_cast<const bf16_t*>(                       \
        &(smem)[(size_t)(mLocal) * (ldk) + (kBase) + (((lane) >> 4) << 3)]);  \
    v8bf _lo = *reinterpret_cast<const v8bf*>(_p);                            \
    v8bf _hi = *reinterpret_cast<const v8bf*>(_p + 16);                       \
    dstv = __builtin_shufflevector(_lo, _hi, 0,1,2,3,4,5,6,7,                 \
                                   8,9,10,11,12,13,14,15);                    \
  } while (0)

// --------------------------- utility kernels --------------------------------

__global__ void k_init(int* bar) {
  if (threadIdx.x < 8 && blockIdx.x == 0) bar[threadIdx.x] = 0;
}

__global__ void k_f2bf(const float* __restrict__ src, unsigned short* __restrict__ dst,
                       int total) {
  int i = blockIdx.x * blockDim.x + threadIdx.x;
  if (i < total) dst[i] = f2bf(src[i]);
}

// W[K][N] fp32 -> WT[N][K] bf16
__global__ void k_transpose(const float* __restrict__ W, unsigned short* __restrict__ WT,
                            int K, int N) {
  int i = blockIdx.x * blockDim.x + threadIdx.x;
  if (i >= K * N) return;
  int k = i / N, n = i % N;
  WT[(size_t)n * K + k] = f2bf(W[i]);
}

// ----------------------- batched X-projection GEMM --------------------------
// XG[g] = xb @ W_x{g}; each wave computes a 32x64 macro tile (2 mT x 4 nT).
// Waves: 3 gates * 512 mG * 16 nG = 24576 -> 3072 blocks of 8 waves.
__global__ void __launch_bounds__(256) k_gemm_xg(const unsigned short* __restrict__ xb,
                                                 const unsigned short* __restrict__ WTx,
                                                 unsigned short* __restrict__ XG) {
  __shared__ alignas(16) unsigned short sA[32 * kI];   // 32KB A slab (shared by WG)
  int wave = blockIdx.x * 8 + (threadIdx.x >> 5);
  int lane = threadIdx.x & 31;
  int gate = wave >> 13;          // 8192 macro tiles per gate
  int rem  = wave & 8191;
  int mG   = rem >> 4;            // [0,512)  (same for all 8 waves of a WG)
  int nG   = rem & 15;            // [0,16)
  int mRow0 = mG * 32;
  const unsigned short* Bm = WTx + (size_t)gate * kH * kI;

  STAGE_TO_LDS(sA, xb + (size_t)mRow0 * kI, (32 * kI * 2) / 16);

  v8f acc[8] = {};
  for (int kb = 0; kb < kI; kb += 32) {
    v16bf a0, a1;
    LOAD_FRAG_A_LDS(a0, sA, kI, (lane & 15),      kb, lane);
    LOAD_FRAG_A_LDS(a1, sA, kI, 16 + (lane & 15), kb, lane);
#pragma unroll
    for (int j = 0; j < 4; ++j) {
      v16bf b = load_frag_b(Bm, kI, (nG * 4 + j) * 16, kb, lane);
      acc[j]     = wmma_bf16(a0, b, acc[j]);
      acc[4 + j] = wmma_bf16(a1, b, acc[4 + j]);
    }
  }
  unsigned short* out = XG + (size_t)gate * kM * kH;
#pragma unroll
  for (int i = 0; i < 2; ++i)
#pragma unroll
    for (int j = 0; j < 4; ++j) {
      int mBase = mRow0 + i * 16 + ((lane >> 4) << 3);
      int n     = (nG * 4 + j) * 16 + (lane & 15);
#pragma unroll
      for (int r = 0; r < 8; ++r)
        out[(size_t)(mBase + r) * kH + n] = f2bf(acc[i * 4 + j][r]);
    }
}

// --------------------------- persistent recurrence --------------------------
__global__ void __launch_bounds__(256, 1)
k_recurrent(const unsigned short* __restrict__ WTh,  // 3 x [kH][kH] bf16 (z,r,h)
            const unsigned short* __restrict__ XG,   // 3 x [kM][kH] bf16
            const float* __restrict__ bz, const float* __restrict__ br,
            const float* __restrict__ bh,
            float* __restrict__ Hf,                  // 2 x [kB][kH] fp32
            unsigned short* __restrict__ Hb,         // 2 x [kB][kH] bf16
            unsigned short* __restrict__ RH,         // [kB][kH] bf16
            float* __restrict__ Zb,                  // [kB][kH] fp32
            unsigned short* __restrict__ Hs,         // [kM][kH] bf16
            int* __restrict__ bar) {
  __shared__ alignas(16) unsigned short sA[16 * kH]; // 32KB A tile
  int tid  = threadIdx.x;
  int lane = tid & 31;
  int wave = blockIdx.x * 8 + (tid >> 5);   // [0,128)
  int gid  = blockIdx.x * 256 + tid;
  int mT   = wave >> 6;                     // [0,2) (uniform within a WG)
  int nT   = wave & 63;                     // [0,64)

  const unsigned short* Wz = WTh;
  const unsigned short* Wr = WTh + (size_t)kH * kH;
  const unsigned short* Wh = WTh + 2ull * kH * kH;
  const unsigned short* XGz = XG;
  const unsigned short* XGr = XG + (size_t)kM * kH;
  const unsigned short* XGh = XG + 2ull * kM * kH;

  for (int i = gid; i < kB * kH; i += kNWG * 256) { Hf[i] = 0.0f; Hb[i] = 0; }
  grid_barrier(bar, bar + 4);

  for (int s = 0; s < kS; ++s) {
    int cur = s & 1, nxt = cur ^ 1;
    const float*          Hfc = Hf + (size_t)cur * kB * kH;
    float*                Hfn = Hf + (size_t)nxt * kB * kH;
    const unsigned short* Hbc = Hb + (size_t)cur * kB * kH;
    unsigned short*       Hbn = Hb + (size_t)nxt * kB * kH;

    // ---- phase A: Z and R gates (shared LDS A tile) ----
    STAGE_TO_LDS(sA, Hbc + (size_t)mT * 16 * kH, (16 * kH * 2) / 16);
    v8f accZ = {}, accR = {};
    for (int kb = 0; kb < kH; kb += 32) {
      v16bf a;
      LOAD_FRAG_A_LDS(a, sA, kH, (lane & 15), kb, lane);
      v16bf bzf = load_frag_b(Wz, kH, nT * 16, kb, lane);
      accZ = wmma_bf16(a, bzf, accZ);
      v16bf brf = load_frag_b(Wr, kH, nT * 16, kb, lane);
      accR = wmma_bf16(a, brf, accR);
    }
    {
      int mBase = mT * 16 + ((lane >> 4) << 3);
      int n     = nT * 16 + (lane & 15);
#pragma unroll
      for (int r = 0; r < 8; ++r) {
        int m = mBase + r;
        size_t gi = ((size_t)m * kS + s) * kH + n;
        float z  = sigmoidf(accZ[r] + bf2f(XGz[gi]) + bz[n]);
        float rv = sigmoidf(accR[r] + bf2f(XGr[gi]) + br[n]);
        Zb[m * kH + n] = z;
        RH[m * kH + n] = f2bf(rv * Hfc[m * kH + n]);
      }
    }
    grid_barrier(bar, bar + 4);

    // ---- phase B: candidate state + GRU update ----
    STAGE_TO_LDS(sA, RH + (size_t)mT * 16 * kH, (16 * kH * 2) / 16);
    v8f accH = {};
    for (int kb = 0; kb < kH; kb += 32) {
      v16bf a;
      LOAD_FRAG_A_LDS(a, sA, kH, (lane & 15), kb, lane);
      v16bf b = load_frag_b(Wh, kH, nT * 16, kb, lane);
      accH = wmma_bf16(a, b, accH);
    }
    {
      int mBase = mT * 16 + ((lane >> 4) << 3);
      int n     = nT * 16 + (lane & 15);
#pragma unroll
      for (int r = 0; r < 8; ++r) {
        int m = mBase + r;
        size_t gi = ((size_t)m * kS + s) * kH + n;
        float ht = tanhf(accH[r] + bf2f(XGh[gi]) + bh[n]);
        float z  = Zb[m * kH + n];
        float h  = Hfc[m * kH + n];
        float hn = z * h + (1.0f - z) * ht;
        Hfn[m * kH + n] = hn;
        unsigned short hb = f2bf(hn);
        Hbn[m * kH + n] = hb;
        Hs[gi] = hb;
      }
    }
    grid_barrier(bar, bar + 4);
  }
}

// --------------------------- output projection ------------------------------
// outputs = Hs @ W_hq + b_q; 2x4 blocked, waves = 512 mG * 8 nG -> 512 blocks.
__global__ void __launch_bounds__(256) k_gemm_y(const unsigned short* __restrict__ Hs,
                                                const unsigned short* __restrict__ WTq,
                                                const float* __restrict__ bq,
                                                float* __restrict__ out) {
  __shared__ alignas(16) unsigned short sA[32 * kH];   // 64KB A slab
  int wave = blockIdx.x * 8 + (threadIdx.x >> 5);
  int lane = threadIdx.x & 31;
  int mG = wave >> 3;   // [0,512)  (uniform within WG)
  int nG = wave & 7;    // [0,8)
  int mRow0 = mG * 32;

  STAGE_TO_LDS(sA, Hs + (size_t)mRow0 * kH, (32 * kH * 2) / 16);

  v8f acc[8] = {};
  for (int kb = 0; kb < kH; kb += 32) {
    v16bf a0, a1;
    LOAD_FRAG_A_LDS(a0, sA, kH, (lane & 15),      kb, lane);
    LOAD_FRAG_A_LDS(a1, sA, kH, 16 + (lane & 15), kb, lane);
#pragma unroll
    for (int j = 0; j < 4; ++j) {
      v16bf b = load_frag_b(WTq, kH, (nG * 4 + j) * 16, kb, lane);
      acc[j]     = wmma_bf16(a0, b, acc[j]);
      acc[4 + j] = wmma_bf16(a1, b, acc[4 + j]);
    }
  }
#pragma unroll
  for (int i = 0; i < 2; ++i)
#pragma unroll
    for (int j = 0; j < 4; ++j) {
      int mBase = mRow0 + i * 16 + ((lane >> 4) << 3);
      int n     = (nG * 4 + j) * 16 + (lane & 15);
#pragma unroll
      for (int r = 0; r < 8; ++r)
        out[(size_t)(mBase + r) * kI + n] = acc[i * 4 + j][r] + bq[n];
    }
}

// ------------------------------- attention ----------------------------------
__global__ void __launch_bounds__(256) k_attn(const unsigned short* __restrict__ Hs,
                                              const float* __restrict__ aW,
                                              const float* __restrict__ ab,
                                              float* __restrict__ outCtx,
                                              float* __restrict__ outW) {
  __shared__ float sw[kS];
  __shared__ float red[2];
  int b = blockIdx.x, tid = threadIdx.x;

  for (int s = tid; s < kS; s += 256) {
    const unsigned short* row = Hs + ((size_t)b * kS + s) * kH;
    float acc = 0.0f;
    for (int k = 0; k < kH; ++k) acc += bf2f(row[k]) * aW[k];
    sw[s] = tanhf(acc + ab[0]);
  }
  __syncthreads();
  if (tid == 0) {
    float mx = -3.4e38f;
    for (int s = 0; s < kS; ++s) mx = fmaxf(mx, sw[s]);
    float sum = 0.0f;
    for (int s = 0; s < kS; ++s) sum += expf(sw[s] - mx);
    red[0] = mx; red[1] = sum;
  }
  __syncthreads();
  float mx = red[0], inv = 1.0f / red[1];
  for (int s = tid; s < kS; s += 256) {
    float w = expf(sw[s] - mx) * inv;
    sw[s] = w;
    outW[(size_t)b * kS + s] = w;
  }
  __syncthreads();
  for (int n = tid; n < kH; n += 256) {
    float acc = 0.0f;
    for (int s = 0; s < kS; ++s)
      acc += sw[s] * bf2f(Hs[((size_t)b * kS + s) * kH + n]);
    outCtx[(size_t)b * kH + n] = acc;
  }
}

// ------------------------------- launcher -----------------------------------

extern "C" void kernel_launch(void* const* d_in, const int* in_sizes, int n_in,
                              void* d_out, int out_size, void* d_ws, size_t ws_size,
                              hipStream_t stream) {
  (void)in_sizes; (void)n_in; (void)out_size; (void)ws_size;

  const float* x    = (const float*)d_in[0];
  const float* W_xz = (const float*)d_in[1];
  const float* W_hz = (const float*)d_in[2];
  const float* b_z  = (const float*)d_in[3];
  const float* W_xr = (const float*)d_in[4];
  const float* W_hr = (const float*)d_in[5];
  const float* b_r  = (const float*)d_in[6];
  const float* W_xh = (const float*)d_in[7];
  const float* W_hh = (const float*)d_in[8];
  const float* b_h  = (const float*)d_in[9];
  const float* W_hq = (const float*)d_in[10];
  const float* b_q  = (const float*)d_in[11];
  const float* aW   = (const float*)d_in[12];
  const float* ab   = (const float*)d_in[13];

  char* ws = (char*)d_ws;
  unsigned short* xb  = (unsigned short*)(ws + OFF_XB);
  unsigned short* WTx = (unsigned short*)(ws + OFF_WTX);
  unsigned short* WTh = (unsigned short*)(ws + OFF_WTH);
  unsigned short* WTq = (unsigned short*)(ws + OFF_WTQ);
  unsigned short* XG  = (unsigned short*)(ws + OFF_XG);
  unsigned short* Hs  = (unsigned short*)(ws + OFF_HS);
  float*          Hf  = (float*)(ws + OFF_HF);
  unsigned short* Hb  = (unsigned short*)(ws + OFF_HB);
  unsigned short* RH  = (unsigned short*)(ws + OFF_RH);
  float*          Zb  = (float*)(ws + OFF_ZB);
  int*            bar = (int*)(ws + OFF_BAR);

  float* out  = (float*)d_out;
  float* oCtx = out + OUT_CTX;
  float* oAtt = out + OUT_ATT;

  k_init<<<1, 64, 0, stream>>>(bar);

  k_f2bf<<<(kM * kI + 255) / 256, 256, 0, stream>>>(x, xb, kM * kI);

  k_transpose<<<(kI * kH + 255) / 256, 256, 0, stream>>>(W_xz, WTx + 0ull * kH * kI, kI, kH);
  k_transpose<<<(kI * kH + 255) / 256, 256, 0, stream>>>(W_xr, WTx + 1ull * kH * kI, kI, kH);
  k_transpose<<<(kI * kH + 255) / 256, 256, 0, stream>>>(W_xh, WTx + 2ull * kH * kI, kI, kH);
  k_transpose<<<(kH * kH + 255) / 256, 256, 0, stream>>>(W_hz, WTh + 0ull * kH * kH, kH, kH);
  k_transpose<<<(kH * kH + 255) / 256, 256, 0, stream>>>(W_hr, WTh + 1ull * kH * kH, kH, kH);
  k_transpose<<<(kH * kH + 255) / 256, 256, 0, stream>>>(W_hh, WTh + 2ull * kH * kH, kH, kH);
  k_transpose<<<(kH * kI + 255) / 256, 256, 0, stream>>>(W_hq, WTq, kH, kI);

  // 3 gates * 512 mG * 16 nG macro tiles, 8 waves/block
  k_gemm_xg<<<(3 * 512 * 16) / 8, 256, 0, stream>>>(xb, WTx, XG);

  k_recurrent<<<kNWG, 256, 0, stream>>>(WTh, XG, b_z, b_r, b_h,
                                        Hf, Hb, RH, Zb, Hs, bar);

  // 512 mG * 8 nG macro tiles
  k_gemm_y<<<(512 * 8) / 8, 256, 0, stream>>>(Hs, WTq, b_q, out);

  k_attn<<<kB, 256, 0, stream>>>(Hs, aW, ab, oCtx, oAtt);
}
